// Layer84_39290360824237
// MI455X (gfx1250) — matrix operand order repaired and checked by reading, not compiled, not run on hardware
//
#include <hip/hip_runtime.h>

typedef __attribute__((ext_vector_type(16))) _Float16 v16h;
typedef __attribute__((ext_vector_type(8)))  _Float16 v8h;
typedef __attribute__((ext_vector_type(8)))  float    v8f;
typedef __attribute__((ext_vector_type(4)))  int      v4i;
typedef __attribute__((ext_vector_type(8)))  int      v8i;
typedef __attribute__((ext_vector_type(4)))  unsigned int v4u;

union Frag { v16h v; v8h h[2]; };

#define B_  16
#define S_  512
#define H_  1024
#define NH_ 16
#define HD_ 64

static __device__ __forceinline__ v8f wmma_f16(const Frag& a, const Frag& b, v8f c) {
  return __builtin_amdgcn_wmma_f32_16x16x32_f16(false, a.v, false, b.v, (short)0, c,
                                                false, false);
}

static __device__ __forceinline__ v8h cvt8(const float4 a, const float4 b) {
  v8h h;
  h[0]=(_Float16)a.x; h[1]=(_Float16)a.y; h[2]=(_Float16)a.z; h[3]=(_Float16)a.w;
  h[4]=(_Float16)b.x; h[5]=(_Float16)b.y; h[6]=(_Float16)b.z; h[7]=(_Float16)b.w;
  return h;
}

// ---- CDNA5 async global->LDS staging (ASYNCcnt-tracked, codegen-verified) --
#if __has_builtin(__builtin_amdgcn_global_load_async_to_lds_b128)
#define HAVE_ASYNC_BUILTIN 1
#endif

static __device__ __forceinline__ void stage_b128(const _Float16* __restrict__ g,
                                                  _Float16* l) {
#if defined(HAVE_ASYNC_BUILTIN)
  __builtin_amdgcn_global_load_async_to_lds_b128(
      (__attribute__((address_space(1))) v4i*)g,
      (__attribute__((address_space(3))) v4i*)l, 0, 0);
#else
  unsigned lds = (unsigned)(unsigned long long)l;
  unsigned long long ga = (unsigned long long)g;
  asm volatile("global_load_async_to_lds_b128 %0, %1, off"
               :: "v"(lds), "v"(ga) : "memory");
#endif
}

#if __has_builtin(__builtin_amdgcn_s_wait_asynccnt)
#define WAIT_ASYNC(n) __builtin_amdgcn_s_wait_asynccnt(n)
#else
#define WAIT_ASYNC(n) asm volatile("s_wait_asynccnt " #n ::: "memory")
#endif

// ---- CDNA5 Tensor Data Mover: one DMA op per 64x32 f16 tile ----------------
#if __has_builtin(__builtin_amdgcn_tensor_load_to_lds)
#define HAVE_TDM 1
#if __has_include(<hip/amd_detail/amd_gfx1250_TDM.h>)
#define TDM_6ARG 1   // amdgpu-toolchain: 6-arg form; ROCm 7.2: 5-arg form
#endif
#endif

#if __has_builtin(__builtin_amdgcn_s_wait_tensorcnt)
#define WAIT_TENSOR(n) __builtin_amdgcn_s_wait_tensorcnt(n)
#else
#define WAIT_TENSOR(n) asm volatile("s_wait_tensorcnt " #n ::: "memory")
#endif

#if defined(HAVE_TDM)
// D# for a 2-D f16 tile: tile_dim0=32 elems (64B rows), tile_dim1=64 rows,
// tensor row stride 1024 elems. LDS pitch 40 halves via pad feature:
// pad_interval code 3 = 16 DWORDs (one 64B row), pad_amount code 3 = 4 DWORDs.
static __device__ __forceinline__ void tdm_load_tile(const _Float16* gbase,
                                                     _Float16* lbase) {
  const unsigned long long ga = (unsigned long long)gbase;
  const unsigned lds = (unsigned)(unsigned long long)lbase;
  v4u g0;
  g0.x = 1u;                                   // count=1 (valid), load, no gather
  g0.y = lds;                                  // lds_addr (bytes)
  g0.z = (unsigned)ga;                         // global_addr[31:0]
  g0.w = (unsigned)(ga >> 32) | (2u << 30);    // global_addr[56:32] | type=2
  v8i g1;
  g1[0] = (1 << 16) | (1 << 20) | (3 << 22) | (3 << 25);
         // data_size=2B | pad_enable | pad_interval=16dw | pad_amount=4dw
  g1[1] = (int)(1024u << 16);                  // tensor_dim0 = 1024
  g1[2] = (int)(8192u << 16);                  // tensor_dim1 = 8192 (no OOB hit)
  g1[3] = (int)(32u << 16);                    // tile_dim0 = 32
  g1[4] = 64;                                  // tile_dim1 = 64
  g1[5] = 1024;                                // tensor_dim0_stride = 1024
  g1[6] = 0;
  g1[7] = 0;
  const v4i z4 = {0, 0, 0, 0};
#if defined(TDM_6ARG)
  const v8i z8 = {0, 0, 0, 0, 0, 0, 0, 0};
  __builtin_amdgcn_tensor_load_to_lds(g0, g1, z4, z4, z8, 0);
#else
  __builtin_amdgcn_tensor_load_to_lds(g0, g1, z4, z4, 0);
#endif
}
#endif

// ---------------------------------------------------------------------------
// Prep A: Xh = f16(X), elementwise. grid 4096 x 256, 8 elems/thread.
// ---------------------------------------------------------------------------
__global__ void __launch_bounds__(256)
cvt_x_kernel(const float* __restrict__ X, _Float16* __restrict__ Xh) {
  const size_t i = ((size_t)blockIdx.x * 256 + threadIdx.x) * 8;
  const float4 a = *(const float4*)(X + i);
  const float4 b = *(const float4*)(X + i + 4);
  *(v8h*)(Xh + i) = cvt8(a, b);
}

// ---------------------------------------------------------------------------
// Prep B: Wt3[z][n][k] = f16(W_z[k][n])  (transpose, LDS-tiled, coalesced).
// grid (16,16,3), block 128. Tile 64x64.
// ---------------------------------------------------------------------------
__global__ void __launch_bounds__(128)
cvt_w_kernel(const float* __restrict__ Wq, const float* __restrict__ Wk,
             const float* __restrict__ Wv, _Float16* __restrict__ Wt3) {
  __shared__ _Float16 Ts[64][72];
  const int z = blockIdx.z;
  const float* W = (z == 0) ? Wq : (z == 1) ? Wk : Wv;
  _Float16* Wt = Wt3 + (size_t)z * H_ * H_;
  const int k0 = blockIdx.x * 64;
  const int n0 = blockIdx.y * 64;
  const int t  = threadIdx.x;
  const int r  = t >> 1;
  const int sg = (t & 1) * 32;

  const float4* p = (const float4*)(W + (size_t)(k0 + r) * H_ + n0 + sg);
  #pragma unroll
  for (int q8 = 0; q8 < 4; ++q8)
    *(v8h*)&Ts[r][sg + q8 * 8] = cvt8(p[2 * q8], p[2 * q8 + 1]);
  __syncthreads();
  #pragma unroll
  for (int q8 = 0; q8 < 4; ++q8) {
    v8h h;
    #pragma unroll
    for (int i = 0; i < 8; ++i) h[i] = Ts[sg + q8 * 8 + i][r];
    *(v8h*)(Wt + (size_t)(n0 + r) * H_ + k0 + sg + q8 * 8) = h;
  }
}

// ---------------------------------------------------------------------------
// Phase 1: Y = (Xh @ Wt^T + bias) * scale, f16 output.
//   z=0 -> Qh [B,NH,S,HD] (1/8 folded)  z=1 -> Kh [B,NH,S,HD]
//   z=2 -> Vt [B,NH,HD,S] (pre-transposed for attention B-fragments)
// grid (128,16,3), block 128 (4 waves). Double-buffered TDM (or async) staging.
// ---------------------------------------------------------------------------
__global__ void __launch_bounds__(128)
qkv_gemm_kernel(const _Float16* __restrict__ Xh, const _Float16* __restrict__ Wt3,
                const float* __restrict__ bq, const float* __restrict__ bk,
                const float* __restrict__ bv,
                _Float16* __restrict__ Qh, _Float16* __restrict__ Kh,
                _Float16* __restrict__ Vt)
{
  __shared__ _Float16 As[2][64][40];   // [m][k] tiles, 80B row pitch
  __shared__ _Float16 Bs[2][64][40];   // [n][k] tiles

  const int z = blockIdx.z;
  const _Float16* Wt   = Wt3 + (size_t)z * H_ * H_;
  const float*    bias = (z == 0) ? bq : (z == 1) ? bk : bv;
  const float scale = (z == 0) ? 0.125f : 1.0f;

  const int m0   = blockIdx.x * 64;
  const int n0   = blockIdx.y * 64;
  const int t    = threadIdx.x;
  const int lane = t & 31;
  const int w    = t >> 5;
  const int half = lane >> 4;
  const int l16  = lane & 15;

  const _Float16* gAbase = Xh + (size_t)m0 * H_;
  const _Float16* gBbase = Wt + (size_t)n0 * H_;

#if defined(HAVE_TDM)
  // Wave 0 drives the Tensor Data Mover; one DMA op per tile.
  if (t < 32) {
    tdm_load_tile(gAbase, &As[0][0][0]);
    tdm_load_tile(gBbase, &Bs[0][0][0]);
  }
#else
  const int r  = t >> 1;
  const int sg = (t & 1) * 16;
  const _Float16* gA = gAbase + (size_t)r * H_ + sg;
  const _Float16* gB = gBbase + (size_t)r * H_ + sg;
  auto issue = [&](int buf, int k0) {
    stage_b128(gA + k0,     &As[buf][r][sg]);
    stage_b128(gA + k0 + 8, &As[buf][r][sg + 8]);
    stage_b128(gB + k0,     &Bs[buf][r][sg]);
    stage_b128(gB + k0 + 8, &Bs[buf][r][sg + 8]);
  };
  issue(0, 0);
#endif

  v8f acc[4] = {};

  for (int kk = 0; kk < 32; ++kk) {
    const int cur = kk & 1;
#if defined(HAVE_TDM)
    if (t < 32) {
      if (kk < 31) {
        tdm_load_tile(gAbase + (kk + 1) * 32, &As[cur ^ 1][0][0]);
        tdm_load_tile(gBbase + (kk + 1) * 32, &Bs[cur ^ 1][0][0]);
        WAIT_TENSOR(2);                // current tile's 2 DMAs retired
      } else {
        WAIT_TENSOR(0);
      }
    }
#else
    if (kk < 31) {
      issue(cur ^ 1, (kk + 1) * 32);
      WAIT_ASYNC(4);
    } else {
      WAIT_ASYNC(0);
    }
#endif
    __syncthreads();                   // publish tile to all waves

    Frag a;
    {
      const int row = w * 16 + l16;
      const int kb  = half ? 8 : 0;    // A layout: K 0-7/16-23 | 8-15/24-31
      a.h[0] = *(const v8h*)&As[cur][row][kb];
      a.h[1] = *(const v8h*)&As[cur][row][kb + 16];
    }
    #pragma unroll
    for (int j = 0; j < 4; ++j) {
      Frag bf;
      const int col = j * 16 + l16;
      const int kb  = half ? 8 : 0;    // B layout: K 0-7/16-23 | 8-15/24-31
      bf.h[0] = *(const v8h*)&Bs[cur][col][kb];
      bf.h[1] = *(const v8h*)&Bs[cur][col][kb + 16];
      acc[j] = wmma_f16(a, bf, acc[j]);
    }
    __syncthreads();                   // readers done before buffer reuse
  }

  #pragma unroll
  for (int j = 0; j < 4; ++j) {
    const int n  = n0 + j * 16 + l16;
    const float bn = bias[n];
    const int hh = n >> 6, d = n & 63;
    #pragma unroll
    for (int i = 0; i < 8; ++i) {
      const int m = m0 + w * 16 + i + half * 8;
      const int bb = m >> 9, s = m & 511;
      const float val = (acc[j][i] + bn) * scale;
      if (z == 2)
        Vt[(((size_t)bb * NH_ + hh) * HD_ + d) * S_ + s] = (_Float16)val;
      else if (z == 1)
        Kh[(((size_t)bb * NH_ + hh) * S_ + s) * HD_ + d] = (_Float16)val;
      else
        Qh[(((size_t)bb * NH_ + hh) * S_ + s) * HD_ + d] = (_Float16)val;
    }
  }
}

// ---------------------------------------------------------------------------
// Phase 2: flash attention, barrier-free inner loop.
// grid (8,16,16) = (S/64, NH, B), block 128 (4 waves, 16 q-rows each).
// ---------------------------------------------------------------------------
__global__ void __launch_bounds__(128)
attn_kernel(const _Float16* __restrict__ Qh, const _Float16* __restrict__ Kh,
            const _Float16* __restrict__ Vt, const float* __restrict__ mask,
            float* __restrict__ out)
{
  __shared__ _Float16 Pb[4][16][64];

  const int b    = blockIdx.z;
  const int hh   = blockIdx.y;
  const int t    = threadIdx.x;
  const int lane = t & 31;
  const int w    = t >> 5;
  const int half = lane >> 4;
  const int l16  = lane & 15;
  const int q0   = blockIdx.x * 64 + w * 16;
  const size_t bh   = ((size_t)b * NH_ + hh) * S_;
  const size_t bh64 = ((size_t)b * NH_ + hh) * HD_;

  Frag aq[2];
  {
    const _Float16* qp = Qh + (bh + q0 + l16) * HD_;
    #pragma unroll
    for (int tt = 0; tt < 2; ++tt) {
      const int off = tt * 32 + (half ? 8 : 0);
      aq[tt].h[0] = *(const v8h*)(qp + off);
      aq[tt].h[1] = *(const v8h*)(qp + off + 16);
    }
  }

  float mrow[8], lrow[8];
  v8f o[4] = {};
  #pragma unroll
  for (int i = 0; i < 8; ++i) { mrow[i] = -1e30f; lrow[i] = 0.0f; }

  for (int kb = 0; kb < 8; ++kb) {
    const int kbase = kb * 64;

    if (kb < 7) {
      __builtin_prefetch(Kh + (bh + kbase + 64 + lane * 2) * HD_, 0, 0);
      __builtin_prefetch(Vt + (bh64 + lane * 2) * S_ + kbase + 64, 0, 0);
    }

    v8f sc[4];
    #pragma unroll
    for (int j = 0; j < 4; ++j) {
      const _Float16* kp = Kh + (bh + kbase + j * 16 + l16) * HD_;
      Frag b0, b1;
      const int off = half ? 8 : 0;
      b0.h[0] = *(const v8h*)(kp + off);
      b0.h[1] = *(const v8h*)(kp + off + 16);
      b1.h[0] = *(const v8h*)(kp + 32 + off);
      b1.h[1] = *(const v8h*)(kp + 32 + off + 16);
      v8f zz = {};
      sc[j] = wmma_f16(aq[0], b0, zz);
      sc[j] = wmma_f16(aq[1], b1, sc[j]);
      const float mv = mask[(size_t)b * S_ + kbase + j * 16 + l16];
      #pragma unroll
      for (int i = 0; i < 8; ++i) sc[j][i] += mv;
    }

    #pragma unroll
    for (int i = 0; i < 8; ++i) {
      float x = fmaxf(fmaxf(sc[0][i], sc[1][i]), fmaxf(sc[2][i], sc[3][i]));
      x = fmaxf(x, __shfl_xor(x, 1, 32));
      x = fmaxf(x, __shfl_xor(x, 2, 32));
      x = fmaxf(x, __shfl_xor(x, 4, 32));
      x = fmaxf(x, __shfl_xor(x, 8, 32));
      const float mn = fmaxf(mrow[i], x);
      const float alpha = __expf(mrow[i] - mn);
      mrow[i] = mn;
      lrow[i] *= alpha;
      #pragma unroll
      for (int n = 0; n < 4; ++n) o[n][i] *= alpha;
    }
    #pragma unroll
    for (int i = 0; i < 8; ++i) {
      float rs = 0.0f;
      #pragma unroll
      for (int j = 0; j < 4; ++j) {
        const float p = __expf(sc[j][i] - mrow[i]);
        sc[j][i] = p;
        rs += p;
      }
      rs += __shfl_xor(rs, 1, 32);
      rs += __shfl_xor(rs, 2, 32);
      rs += __shfl_xor(rs, 4, 32);
      rs += __shfl_xor(rs, 8, 32);
      lrow[i] += rs;
    }

    #pragma unroll
    for (int j = 0; j < 4; ++j)
      #pragma unroll
      for (int i = 0; i < 8; ++i)
        Pb[w][i + half * 8][j * 16 + l16] = (_Float16)sc[j][i];
    asm volatile("s_wait_dscnt 0x0" ::: "memory");

    Frag pa[2];
    #pragma unroll
    for (int tt = 0; tt < 2; ++tt) {
      const int off = tt * 32 + (half ? 8 : 0);
      pa[tt].h[0] = *(const v8h*)&Pb[w][l16][off];
      pa[tt].h[1] = *(const v8h*)&Pb[w][l16][off + 16];
    }

    #pragma unroll
    for (int n = 0; n < 4; ++n) {
      const _Float16* vp = Vt + (bh64 + n * 16 + l16) * S_ + kbase;
      const int off = half ? 8 : 0;
      Frag b0, b1;
      b0.h[0] = *(const v8h*)(vp + off);
      b0.h[1] = *(const v8h*)(vp + off + 16);
      b1.h[0] = *(const v8h*)(vp + 32 + off);
      b1.h[1] = *(const v8h*)(vp + 32 + off + 16);
      o[n] = wmma_f16(pa[0], b0, o[n]);
      o[n] = wmma_f16(pa[1], b1, o[n]);
    }
  }

  #pragma unroll
  for (int n = 0; n < 4; ++n) {
    #pragma unroll
    for (int i = 0; i < 8; ++i) {
      const int q = q0 + i + half * 8;
      out[((size_t)b * S_ + q) * H_ + hh * HD_ + n * 16 + l16] = o[n][i] / lrow[i];
    }
  }
}

// ---------------------------------------------------------------------------
extern "C" void kernel_launch(void* const* d_in, const int* in_sizes, int n_in,
                              void* d_out, int out_size, void* d_ws, size_t ws_size,
                              hipStream_t stream) {
  const float* mask = (const float*)d_in[0];
  const float* X    = (const float*)d_in[1];
  const float* Wq   = (const float*)d_in[2];
  const float* bq   = (const float*)d_in[3];
  const float* Wk   = (const float*)d_in[4];
  const float* bk   = (const float*)d_in[5];
  const float* Wv   = (const float*)d_in[6];
  const float* bv   = (const float*)d_in[7];
  float* out = (float*)d_out;

  const size_t n_qkv = (size_t)B_ * NH_ * S_ * HD_;
  _Float16* Qh  = (_Float16*)d_ws;
  _Float16* Kh  = Qh + n_qkv;
  _Float16* Vt  = Kh + n_qkv;
  _Float16* Xh  = Vt + n_qkv;
  _Float16* Wt3 = Xh + n_qkv;

  cvt_x_kernel<<<dim3(4096), dim3(256), 0, stream>>>(X, Xh);
  cvt_w_kernel<<<dim3(16, 16, 3), dim3(128), 0, stream>>>(Wq, Wk, Wv, Wt3);
  qkv_gemm_kernel<<<dim3(128, 16, 3), dim3(128), 0, stream>>>(
      Xh, Wt3, bq, bk, bv, Qh, Kh, Vt);
  attn_kernel<<<dim3(8, 16, 16), dim3(128), 0, stream>>>(
      Qh, Kh, Vt, mask, out);
}
